// CaptionDecoder_21311627722950
// MI455X (gfx1250) — compile-verified, matrix operand order checked by reading, not run on hardware
//
#include <hip/hip_runtime.h>
#include <hip/hip_bf16.h>

typedef __attribute__((ext_vector_type(16))) _Float16 v16h;
typedef __attribute__((ext_vector_type(8)))  _Float16 h8;
typedef __attribute__((ext_vector_type(8)))  float    v8f;

#define VOCAB  32000
#define EMBED  512
#define HIDDEN 512
#define CTX    2048
#define BB     32
#define SS     64
#define G4     2048   // 4*HIDDEN
#define MROWS  2048   // B*S

__device__ __forceinline__ float sigmoidf_(float x) {
    return 1.0f / (1.0f + __expf(-x));
}

// ---------------------------------------------------------------------------
// f32 -> f16 conversion (grid-stride)
// ---------------------------------------------------------------------------
__global__ void cvt_f32_to_f16_kernel(const float* __restrict__ src,
                                      _Float16* __restrict__ dst, int n) {
    for (int i = blockIdx.x * blockDim.x + threadIdx.x; i < n;
         i += gridDim.x * blockDim.x) {
        dst[i] = (_Float16)src[i];
    }
}

// ---------------------------------------------------------------------------
// Embedding gather -> f16 rows (row = b*S + s)
// ---------------------------------------------------------------------------
__global__ void gather_emb_kernel(const int* __restrict__ tokens,
                                  const float* __restrict__ table,
                                  _Float16* __restrict__ out) {
    int idx = blockIdx.x * blockDim.x + threadIdx.x;  // [0, MROWS*EMBED)
    if (idx >= MROWS * EMBED) return;
    int row = idx >> 9;          // /512
    int k   = idx & 511;
    int t   = tokens[row];
    out[idx] = (_Float16)table[(size_t)t * EMBED + k];
}

// ---------------------------------------------------------------------------
// h0 = tanh(context @ W_ih + b_ih)  (stored f16 for WMMA A-operand)
// c0 = tanh(context @ W_ic + b_ic)  (stored f32)
// ---------------------------------------------------------------------------
__global__ void init_state_kernel(const float* __restrict__ context,
                                  const float* __restrict__ W_ih,
                                  const float* __restrict__ b_ih,
                                  const float* __restrict__ W_ic,
                                  const float* __restrict__ b_ic,
                                  _Float16* __restrict__ h,
                                  float* __restrict__ c) {
    int idx = blockIdx.x * blockDim.x + threadIdx.x;  // [0, BB*HIDDEN)
    if (idx >= BB * HIDDEN) return;
    int bi = idx >> 9;
    int j  = idx & 511;
    const float* ctx = context + (size_t)bi * CTX;
    float ah = 0.f, ac = 0.f;
    for (int k = 0; k < CTX; ++k) {
        float cv = ctx[k];
        ah = fmaf(cv, W_ih[(size_t)k * HIDDEN + j], ah);
        ac = fmaf(cv, W_ic[(size_t)k * HIDDEN + j], ac);
    }
    h[idx] = (_Float16)tanhf(ah + b_ih[j]);
    c[idx] = tanhf(ac + b_ic[j]);
}

// ---------------------------------------------------------------------------
// WMMA GEMM:  C[M,N] (f32) = A[M,K] (f16) * B[K,N] (f16) (+ bias[N])
// Block = 256 threads = 8 wave32 waves, arranged 2(M) x 4(N).
// Block tile BM=32, BN=256, BK=32. Each wave: 16(M) x 64(N) = 4 WMMA tiles.
// LDS double-buffered; software pipeline overlaps next-tile global loads
// with current-tile WMMAs:
//   frag ds_loads(buf) -> global loads(k+1) -> 4x WMMA -> ds_stores(buf^1)
//   -> one __syncthreads() per iteration.
// ---------------------------------------------------------------------------
__global__ __launch_bounds__(256)
void wmma_gemm_f16_kernel(const _Float16* __restrict__ A, int lda,
                          const _Float16* __restrict__ Bm, int ldb,
                          const float* __restrict__ bias,
                          float* __restrict__ C, int ldc, int K) {
    __shared__ __align__(16) _Float16 Ah[2][32][40];    // 32(M) x 32(K), padded
    __shared__ __align__(16) _Float16 Bt[2][256][40];   // 256(N) x 32(K), padded

    const int tid  = threadIdx.x;
    const int lane = tid & 31;
    const int w    = tid >> 5;
    const int wm   = w & 1;        // M 16-tile within block
    const int wn   = w >> 1;       // N 64-slab within block (0..3)
    const int m0   = blockIdx.x * 32;
    const int n0   = blockIdx.y * 256;

    v8f acc[4] = {{}, {}, {}, {}};

    // staging indices
    const int ar = tid >> 3;            // A row 0..31
    const int ak = (tid & 7) * 4;       // A col group (4 halves)
    const int nb = n0 + tid;            // B column owned by this thread

    // fragment indices (CDNA5 16-bit layout: lanes 0-15 / 16-31 split K halves)
    const int fr = lane & 15;
    const int kb = (lane >> 4) << 3;    // 0 or 8

    // ---- prologue: stage tile k0=0 into buffer 0 ----
    {
        const _Float16* asrc = A + (size_t)(m0 + ar) * lda + ak;
        Ah[0][ar][ak + 0] = asrc[0];
        Ah[0][ar][ak + 1] = asrc[1];
        Ah[0][ar][ak + 2] = asrc[2];
        Ah[0][ar][ak + 3] = asrc[3];
        #pragma unroll
        for (int kk = 0; kk < 32; kk += 8) {
            h8 v;
            #pragma unroll
            for (int j = 0; j < 8; ++j)
                v[j] = Bm[(size_t)(kk + j) * ldb + nb];   // SGPR row base + v-offset
            *(h8*)&Bt[0][tid][kk] = v;
        }
    }
    __syncthreads();

    int buf = 0;
    for (int k0 = 0; k0 < K; k0 += 32, buf ^= 1) {
        // ---- (a) fragment loads from current buffer ----
        const _Float16* arow = &Ah[buf][wm * 16 + fr][0];
        h8 alo = *(const h8*)(arow + kb);
        h8 ahi = *(const h8*)(arow + 16 + kb);
        v16h af = __builtin_shufflevector(alo, ahi,
                    0,1,2,3,4,5,6,7,8,9,10,11,12,13,14,15);
        v16h bf[4];
        #pragma unroll
        for (int n = 0; n < 4; ++n) {
            const _Float16* bcol = &Bt[buf][wn * 64 + n * 16 + fr][0];
            h8 blo = *(const h8*)(bcol + kb);
            h8 bhi = *(const h8*)(bcol + 16 + kb);
            bf[n] = __builtin_shufflevector(blo, bhi,
                        0,1,2,3,4,5,6,7,8,9,10,11,12,13,14,15);
        }

        // ---- (b) next tile's global loads into registers ----
        const bool more = (k0 + 32) < K;
        _Float16 areg[4];
        _Float16 breg[32];
        if (more) {
            const int kn = k0 + 32;
            const _Float16* asrc = A + (size_t)(m0 + ar) * lda + kn + ak;
            #pragma unroll
            for (int j = 0; j < 4; ++j) areg[j] = asrc[j];
            #pragma unroll
            for (int j = 0; j < 32; ++j)
                breg[j] = Bm[(size_t)(kn + j) * ldb + nb];
        }

        // ---- (c) 4 back-to-back WMMAs (independent accumulators) ----
        #pragma unroll
        for (int n = 0; n < 4; ++n) {
            acc[n] = __builtin_amdgcn_wmma_f32_16x16x32_f16(
                false, af, false, bf[n], (short)0, acc[n], false, false);
        }

        // ---- (d) store next tile into the other buffer ----
        if (more) {
            const int nbuf = buf ^ 1;
            Ah[nbuf][ar][ak + 0] = areg[0];
            Ah[nbuf][ar][ak + 1] = areg[1];
            Ah[nbuf][ar][ak + 2] = areg[2];
            Ah[nbuf][ar][ak + 3] = areg[3];
            #pragma unroll
            for (int kk = 0; kk < 32; kk += 8) {
                h8 v;
                #pragma unroll
                for (int j = 0; j < 8; ++j) v[j] = breg[kk + j];
                *(h8*)&Bt[nbuf][tid][kk] = v;
            }
        }
        // ---- (e) one barrier per iteration ----
        __syncthreads();
    }

    // ---- store: C/D layout — VGPR r: lanes 0-15 -> M=r, lanes 16-31 -> M=8+r
    const int mh = (lane >> 4) << 3;
    #pragma unroll
    for (int n = 0; n < 4; ++n) {
        const int ncol = n0 + wn * 64 + n * 16 + fr;
        const float bv = bias ? bias[ncol] : 0.0f;
        #pragma unroll
        for (int r = 0; r < 8; ++r) {
            const int m = m0 + wm * 16 + mh + r;
            C[(size_t)m * ldc + ncol] = acc[n][r] + bv;
        }
    }
}

// ---------------------------------------------------------------------------
// LSTM gate/update for step t.
// ---------------------------------------------------------------------------
__global__ void lstm_gate_kernel(const float* __restrict__ z,
                                 const float* __restrict__ xg,
                                 const int* __restrict__ tokens,
                                 _Float16* __restrict__ h,
                                 float* __restrict__ c,
                                 _Float16* __restrict__ hs,
                                 int t) {
    int idx = blockIdx.x * blockDim.x + threadIdx.x;  // [0, BB*HIDDEN)
    if (idx >= BB * HIDDEN) return;
    int bi = idx >> 9;
    int j  = idx & 511;
    const float* zr  = z  + (size_t)bi * G4;
    const float* xgr = xg + ((size_t)bi * SS + t) * G4;

    float zi = zr[j]        + xgr[j];
    float zf = zr[512 + j]  + xgr[512 + j];
    float zg = zr[1024 + j] + xgr[1024 + j];
    float zo = zr[1536 + j] + xgr[1536 + j];

    float cv    = c[idx];
    float c_new = sigmoidf_(zf) * cv + sigmoidf_(zi) * tanhf(zg);
    float h_new = sigmoidf_(zo) * tanhf(c_new);

    bool  m     = tokens[bi * SS + t] != 0;
    float h_out = m ? h_new : (float)h[idx];
    float c_out = m ? c_new : cv;

    h[idx] = (_Float16)h_out;
    c[idx] = c_out;
    hs[((size_t)bi * SS + t) * HIDDEN + j] = (_Float16)h_out;
}

// ---------------------------------------------------------------------------
// Host launcher
// ---------------------------------------------------------------------------
extern "C" void kernel_launch(void* const* d_in, const int* in_sizes, int n_in,
                              void* d_out, int out_size, void* d_ws, size_t ws_size,
                              hipStream_t stream) {
    const int*   tokens    = (const int*)  d_in[0];
    const float* context   = (const float*)d_in[1];
    const float* emb_table = (const float*)d_in[2];
    const float* W_ih      = (const float*)d_in[3];
    const float* b_ih      = (const float*)d_in[4];
    const float* W_ic      = (const float*)d_in[5];
    const float* b_ic      = (const float*)d_in[6];
    const float* W_x       = (const float*)d_in[7];
    const float* W_h       = (const float*)d_in[8];
    const float* bvec      = (const float*)d_in[9];
    const float* W_out     = (const float*)d_in[10];
    const float* b_out     = (const float*)d_in[11];
    float* out = (float*)d_out;

    // workspace carve-up
    char* p = (char*)d_ws;
    _Float16* Wx_h   = (_Float16*)p; p += (size_t)EMBED  * G4    * 2;   // 2 MB
    _Float16* Wh_h   = (_Float16*)p; p += (size_t)HIDDEN * G4    * 2;   // 2 MB
    _Float16* Wout_h = (_Float16*)p; p += (size_t)HIDDEN * VOCAB * 2;   // 32 MB
    _Float16* emb_h  = (_Float16*)p; p += (size_t)MROWS  * EMBED * 2;   // 2 MB
    _Float16* hs_h   = (_Float16*)p; p += (size_t)MROWS  * HIDDEN* 2;   // 2 MB
    _Float16* h_st   = (_Float16*)p; p += (size_t)BB * HIDDEN * 2;      // 32 KB
    float*    xg     = (float*)p;    p += (size_t)MROWS * G4 * 4;       // 16 MB
    float*    c_st   = (float*)p;    p += (size_t)BB * HIDDEN * 4;      // 64 KB
    float*    z_buf  = (float*)p;    p += (size_t)BB * G4 * 4;          // 256 KB

    // 1. convert weights to f16
    cvt_f32_to_f16_kernel<<<2048, 256, 0, stream>>>(W_x,   Wx_h,   EMBED  * G4);
    cvt_f32_to_f16_kernel<<<2048, 256, 0, stream>>>(W_h,   Wh_h,   HIDDEN * G4);
    cvt_f32_to_f16_kernel<<<4096, 256, 0, stream>>>(W_out, Wout_h, HIDDEN * VOCAB);

    // 2. embedding gather (f16 rows)
    gather_emb_kernel<<<(MROWS * EMBED) / 256, 256, 0, stream>>>(tokens, emb_table, emb_h);

    // 3. initial h/c
    init_state_kernel<<<(BB * HIDDEN) / 256, 256, 0, stream>>>(
        context, W_ih, b_ih, W_ic, b_ic, h_st, c_st);

    // 4. xg = emb @ W_x + b    (2048 x 2048, K=512)
    {
        dim3 grid(MROWS / 32, G4 / 256);
        wmma_gemm_f16_kernel<<<grid, 256, 0, stream>>>(
            emb_h, EMBED, Wx_h, G4, bvec, xg, G4, EMBED);
    }

    // 5. LSTM recurrence: 64 sequential steps
    for (int t = 0; t < SS; ++t) {
        dim3 gridz(BB / 32, G4 / 256);       // z = h @ W_h  (32 x 2048, K=512)
        wmma_gemm_f16_kernel<<<gridz, 256, 0, stream>>>(
            h_st, HIDDEN, Wh_h, G4, nullptr, z_buf, G4, HIDDEN);
        lstm_gate_kernel<<<(BB * HIDDEN) / 256, 256, 0, stream>>>(
            z_buf, xg, tokens, h_st, c_st, hs_h, t);
    }

    // 6. logits = hs @ W_out + b_out   (2048 x 32000, K=512)
    {
        dim3 grid(MROWS / 32, VOCAB / 256);
        wmma_gemm_f16_kernel<<<grid, 256, 0, stream>>>(
            hs_h, HIDDEN, Wout_h, VOCAB, b_out, out, VOCAB, HIDDEN);
    }
}
